// WindowAttention_4243427688668
// MI455X (gfx1250) — matrix operand order repaired and checked by reading, not compiled, not run on hardware
//
#include <hip/hip_runtime.h>
#include <hip/hip_bf16.h>

typedef __attribute__((ext_vector_type(16))) _Float16 v16h;
typedef __attribute__((ext_vector_type(8)))  _Float16 v8h;
typedef __attribute__((ext_vector_type(8)))  float    v8f;
typedef int b128_t __attribute__((vector_size(16)));   // matches builtin param type

#define WMMA_F16(a, b, c) \
    __builtin_amdgcn_wmma_f32_16x16x32_f16(false, (a), false, (b), (short)0, (c), false, false)

// ---- gfx1250 async global->LDS path (guarded; falls back to plain LDS stores)
#if defined(__has_builtin)
#if __has_builtin(__builtin_amdgcn_global_load_async_to_lds_b128)
#define ASYNC_LDS 1
#endif
#endif

__device__ __forceinline__ void lds_copy16(_Float16* dst, const _Float16* src) {
#ifdef ASYNC_LDS
    __builtin_amdgcn_global_load_async_to_lds_b128(
        (__attribute__((address_space(1))) b128_t*)(unsigned long long)src,
        (__attribute__((address_space(3))) b128_t*)(unsigned)(unsigned long long)(void*)dst,
        0, 0);
#else
    *(v8h*)dst = *(const v8h*)src;
#endif
}

__device__ __forceinline__ void async_wait_all() {
#ifdef ASYNC_LDS
#if __has_builtin(__builtin_amdgcn_s_wait_asynccnt)
    __builtin_amdgcn_s_wait_asynccnt(0);
#else
    asm volatile("s_wait_asynccnt 0" ::: "memory");
#endif
#endif
}

// ---------------- problem constants (fixed by reference) ----------------
static constexpr int  kBATCH = 4;
static constexpr int  kGH = 128, kGW = 128;
static constexpr int  kC  = 1024;          // model dim
static constexpr int  kH  = 16;            // heads
static constexpr int  kD  = 64;            // head dim
static constexpr int  kT  = 256;           // tokens per 16x16 window
static constexpr long long kNTOK = (long long)kGH * kGW;        // 16384 per image
static constexpr long long kM    = (long long)kBATCH * kNTOK;   // 65536 tokens total
static constexpr int  kNQKV = 3 * kC;      // 3072
// workspace element counts (f16)
static constexpr long long X16_N    = kM * kC;        // 67,108,864
static constexpr long long WQKV_N   = (long long)kNQKV * kC;
static constexpr long long WPROJ_N  = (long long)kC * kC;
static constexpr long long QKV_N    = kM * kNQKV;     // 201,326,592
static constexpr long long O16_N    = kM * kC;

// ---------------- f32 -> f16 convert ----------------
__global__ void cvt_f32_to_f16(const float* __restrict__ in,
                               _Float16* __restrict__ out, long long n) {
    long long i = (long long)blockIdx.x * blockDim.x + threadIdx.x;
    long long stride = (long long)gridDim.x * blockDim.x;
    for (; i < n; i += stride) out[i] = (_Float16)in[i];
}

// ---------------- WMMA GEMM: Cout[m,n] = sum_k A[m,k] * Bw[n,k] ----------------
// A: [M,K] f16 row-major; Bw: [N,K] f16 row-major (i.e. B already transposed).
// Block = 256 threads (8 waves), tile 128(M) x 64(N); wave -> 32x32 (2x2 frags).
// Block tile staged into double-buffered LDS via async global->LDS DMA; the
// stage for step s+1 is issued before the WMMAs of step s (ASYNCcnt overlap).
template <bool F16OUT>
__global__ __launch_bounds__(256)
void gemm_wmma(const _Float16* __restrict__ A, const _Float16* __restrict__ Bw,
               void* __restrict__ Cout, int Mdim, int Ndim, int Kdim) {
    __shared__ _Float16 As[2][128 * 32];   // 16 KB
    __shared__ _Float16 Bs[2][64 * 32];    //  8 KB
    const int tid   = threadIdx.x;
    const int wave  = tid >> 5;
    const int lane  = tid & 31;
    const int wm    = wave & 3;            // 4 waves along M
    const int wn    = wave >> 2;           // 2 waves along N
    const int mtile = blockIdx.x * 128;
    const int ntile = blockIdx.y * 64;
    const int lrow  = lane & 15;
    const int ahalf = (lane >> 4) << 4;    // A frag: K base {0,16}
    const int bhalf = (lane >> 4) << 4;    // B frag: K base {0,16}

    auto stage = [&](int buf, int k0) {
        // A tile: 128 rows x 32 halves = 512 16B chunks, 2 per thread
#pragma unroll
        for (int c = 0; c < 2; ++c) {
            const int chunk = tid * 2 + c;
            const int row = chunk >> 2, off = (chunk & 3) * 8;
            lds_copy16(&As[buf][row * 32 + off],
                       A + (size_t)(mtile + row) * Kdim + k0 + off);
        }
        // B tile: 64 rows x 32 halves = 256 16B chunks, 1 per thread
        {
            const int row = tid >> 2, off = (tid & 3) * 8;
            lds_copy16(&Bs[buf][row * 32 + off],
                       Bw + (size_t)(ntile + row) * Kdim + k0 + off);
        }
    };

    v8f acc[2][2] = {};
    stage(0, 0);
    const int nsteps = Kdim >> 5;
    for (int s = 0; s < nsteps; ++s) {
        async_wait_all();          // our copies for buf s are done
        __syncthreads();           // everyone's copies done; prev reads done
        if (s + 1 < nsteps) stage((s + 1) & 1, (s + 1) << 5);
        const int buf = s & 1;
        v16h a[2], b[2];
#pragma unroll
        for (int i = 0; i < 2; ++i) {
            // A fragment 16x32: lane holds row m, K = ahalf + 0..15 (contiguous)
            const _Float16* ap = &As[buf][(wm * 32 + i * 16 + lrow) * 32 + ahalf];
            v8h alo = *(const v8h*)ap;
            v8h ahi = *(const v8h*)(ap + 8);
#pragma unroll
            for (int e = 0; e < 8; ++e) { a[i][e] = alo[e]; a[i][e + 8] = ahi[e]; }
            // B fragment 32x16: lane holds col n, K = bhalf + 0..15 (contiguous)
            const _Float16* bp = &Bs[buf][(wn * 32 + i * 16 + lrow) * 32 + bhalf];
            v8h blo = *(const v8h*)bp;
            v8h bhi = *(const v8h*)(bp + 8);
#pragma unroll
            for (int e = 0; e < 8; ++e) { b[i][e] = blo[e]; b[i][e + 8] = bhi[e]; }
        }
#pragma unroll
        for (int i = 0; i < 2; ++i)
#pragma unroll
            for (int j = 0; j < 2; ++j)
                acc[i][j] = WMMA_F16(a[i], b[j], acc[i][j]);
        __syncthreads();           // all reads of buf done before it is rewritten
    }
    // C/D layout: VGPR r, lane l -> M = r + 8*(l>>4), N = l&15
    const int roff = (lane >> 4) << 3;
    const int mblk = mtile + wm * 32;
    const int nblk = ntile + wn * 32;
#pragma unroll
    for (int i = 0; i < 2; ++i)
#pragma unroll
        for (int j = 0; j < 2; ++j)
#pragma unroll
            for (int r = 0; r < 8; ++r) {
                size_t row = (size_t)(mblk + i * 16 + r + roff);
                size_t col = (size_t)(nblk + j * 16 + lrow);
                if (F16OUT)
                    ((_Float16*)Cout)[row * Ndim + col] = (_Float16)acc[i][j][r];
                else
                    ((float*)Cout)[row * Ndim + col] = acc[i][j][r];
            }
}

// ---------------- RoPE on q,k in place (original token order) ----------------
// thread per (token, pair p in 0..31); applies to all 16 heads for q and k.
__global__ void rope_kernel(_Float16* __restrict__ qkv) {
    long long idx = (long long)blockIdx.x * blockDim.x + threadIdx.x;
    if (idx >= kM * 32) return;
    const int p = (int)(idx & 31);
    const long long t = idx >> 5;
    const int pos = (int)(t % kNTOK);
    const int row = pos >> 7, col = pos & 127;
    const float ph = (float)(row & 15), pw = (float)(col & 15);
    // inv_freq[p] = 10000^(-p/32); freqs = concat(ph*inv[0:16], pw*inv[16:32])
    const float invf = __expf(-(float)p * (9.210340371976184f / 32.0f));
    const float f = (p < 16 ? ph : pw) * invf;
    const float c = __cosf(f), s = __sinf(f);
    _Float16* base = qkv + t * kNQKV;
#pragma unroll
    for (int h = 0; h < kH; ++h) {
#pragma unroll
        for (int part = 0; part < 2; ++part) {      // q then k
            _Float16* e = base + part * kC + h * kD + 2 * p;
            float x1 = (float)e[0], x2 = (float)e[1];
            e[0] = (_Float16)(x1 * c - x2 * s);
            e[1] = (_Float16)(x1 * s + x2 * c);
        }
    }
}

// ---------------- flash window attention ----------------
// grid = (16 heads, 256 windows); block = 512 (16 waves, one 16-row q tile each)
__global__ __launch_bounds__(512)
void attn_kernel(const _Float16* __restrict__ qkv, _Float16* __restrict__ Og) {
    __shared__ _Float16 ksm[kT * kD];        // 32 KB
    __shared__ _Float16 vsm[kT * kD];        // 32 KB
    __shared__ _Float16 psm[16][16 * 32];    // 16 KB, per-wave P staging
    const int head = blockIdx.x;
    const int win  = blockIdx.y;
    const int img  = win >> 6;
    const int w    = win & 63;
    const int wr0  = (w >> 3) << 4;
    const int wc0  = (w & 7) << 4;
    const int tid  = threadIdx.x;

    auto orig = [&](int t) -> long long {  // window token -> original token index
        int r = wr0 + (t >> 4), c = wc0 + (t & 15);
        return (long long)img * kNTOK + (long long)r * kGW + c;
    };

    // async gather of K,V into LDS (8 halves / 16B per chunk)
    for (int chunk = tid; chunk < kT * kD / 8; chunk += 512) {
        const int t  = chunk >> 3;
        const int d0 = (chunk & 7) * 8;
        const long long gb = orig(t) * kNQKV + (long long)head * kD + d0;
        lds_copy16(ksm + t * kD + d0, qkv + gb + kC);
        lds_copy16(vsm + t * kD + d0, qkv + gb + 2 * kC);
    }
    async_wait_all();
    __syncthreads();

    const int wave  = tid >> 5, lane = tid & 31;
    const int lrow  = lane & 15;
    const int ahalf = (lane >> 4) << 4;
    const int bhalf = (lane >> 4) << 4;
    const int mbase = wave * 16;

    // q A-fragments (K = head dim, two 16x32 frags cover d=0..63)
    v16h aq[2];
    {
        const long long qb = orig(mbase + lrow) * kNQKV + (long long)head * kD;
#pragma unroll
        for (int j = 0; j < 2; ++j) {
            const _Float16* p = qkv + qb + j * 32 + ahalf;
            v8h lo = *(const v8h*)p;
            v8h hi = *(const v8h*)(p + 8);
#pragma unroll
            for (int e = 0; e < 8; ++e) { aq[j][e] = lo[e]; aq[j][e + 8] = hi[e]; }
        }
    }

    const float scale = 0.125f;   // 64^-0.5
    float rm[8], rl[8];
#pragma unroll
    for (int r = 0; r < 8; ++r) { rm[r] = -1e30f; rl[r] = 0.0f; }
    v8f o[4] = {};

    for (int kt = 0; kt < 8; ++kt) {       // 32 keys per iteration
        v8f s[2];
#pragma unroll
        for (int sb = 0; sb < 2; ++sb) {
            const int key = kt * 32 + sb * 16 + lrow;
            v8f c = {};
#pragma unroll
            for (int j = 0; j < 2; ++j) {  // K(=d) steps of 32
                // B frag from k: lane holds key column, K(d) = j*32 + bhalf + 0..15
                const _Float16* p = ksm + key * kD + j * 32 + bhalf;
                v8h lo = *(const v8h*)p;
                v8h hi = *(const v8h*)(p + 8);
                v16h bk;
#pragma unroll
                for (int e = 0; e < 8; ++e) { bk[e] = lo[e]; bk[e + 8] = hi[e]; }
                c = WMMA_F16(aq[j], bk, c);
            }
#pragma unroll
            for (int r = 0; r < 8; ++r) s[sb][r] = c[r] * scale;
        }
        // online softmax over the 32 new keys (rows live in half-waves)
        float p0[8], p1[8];
#pragma unroll
        for (int r = 0; r < 8; ++r) {
            float mx = fmaxf(s[0][r], s[1][r]);
#pragma unroll
            for (int msk = 1; msk < 16; msk <<= 1) mx = fmaxf(mx, __shfl_xor(mx, msk, 32));
            mx = fmaxf(mx, rm[r]);
            const float alpha = __expf(rm[r] - mx);
            rm[r] = mx;
            p0[r] = __expf(s[0][r] - mx);
            p1[r] = __expf(s[1][r] - mx);
            float ls = p0[r] + p1[r];
#pragma unroll
            for (int msk = 1; msk < 16; msk <<= 1) ls += __shfl_xor(ls, msk, 32);
            rl[r] = rl[r] * alpha + ls;
#pragma unroll
            for (int f = 0; f < 4; ++f) o[f][r] *= alpha;
        }
        // stage P in LDS to convert C-layout -> A-layout
#pragma unroll
        for (int r = 0; r < 8; ++r) {
            const int row = r + ((lane >> 4) << 3);
            psm[wave][row * 32 + lrow]      = (_Float16)p0[r];
            psm[wave][row * 32 + 16 + lrow] = (_Float16)p1[r];
        }
        asm volatile("s_wait_dscnt 0" ::: "memory");
        v16h ap;
        {
            const _Float16* p = &psm[wave][lrow * 32 + ahalf];
            v8h lo = *(const v8h*)p;
            v8h hi = *(const v8h*)(p + 8);
#pragma unroll
            for (int e = 0; e < 8; ++e) { ap[e] = lo[e]; ap[e + 8] = hi[e]; }
        }
        // O += P @ V  (4 n-tiles of 16 over head dim)
#pragma unroll
        for (int f = 0; f < 4; ++f) {
            v16h bv;
#pragma unroll
            for (int e = 0; e < 16; ++e)
                bv[e] = vsm[(kt * 32 + bhalf + e) * kD + f * 16 + lrow];
            o[f] = WMMA_F16(ap, bv, o[f]);
        }
    }
    // normalize + scatter back to original token order
#pragma unroll
    for (int r = 0; r < 8; ++r) {
        const int t = mbase + r + ((lane >> 4) << 3);
        const long long ob = orig(t) * kC + (long long)head * kD;
        const float inv = 1.0f / rl[r];
#pragma unroll
        for (int f = 0; f < 4; ++f)
            Og[ob + f * 16 + lrow] = (_Float16)(o[f][r] * inv);
    }
}

// ---------------- host-side launch ----------------
extern "C" void kernel_launch(void* const* d_in, const int* in_sizes, int n_in,
                              void* d_out, int out_size, void* d_ws, size_t ws_size,
                              hipStream_t stream) {
    const float* x      = (const float*)d_in[0];
    // d_in[1] = position_ids (unused: positions derived from token index)
    const float* W_qkv  = (const float*)d_in[2];
    const float* W_proj = (const float*)d_in[3];
    float* out = (float*)d_out;

    _Float16* x16     = (_Float16*)d_ws;
    _Float16* wqkv16  = x16 + X16_N;
    _Float16* wproj16 = wqkv16 + WQKV_N;
    _Float16* qkv16   = wproj16 + WPROJ_N;
    _Float16* o16     = qkv16 + QKV_N;

    // 1) convert inputs to f16
    cvt_f32_to_f16<<<(int)((X16_N + 255) / 256), 256, 0, stream>>>(x, x16, X16_N);
    cvt_f32_to_f16<<<(int)((WQKV_N + 255) / 256), 256, 0, stream>>>(W_qkv, wqkv16, WQKV_N);
    cvt_f32_to_f16<<<(int)((WPROJ_N + 255) / 256), 256, 0, stream>>>(W_proj, wproj16, WPROJ_N);

    // 2) QKV projection: [65536,1024] x [3072,1024]^T -> [65536,3072] f16
    gemm_wmma<true><<<dim3((int)(kM / 128), kNQKV / 64), 256, 0, stream>>>(
        x16, wqkv16, (void*)qkv16, (int)kM, kNQKV, kC);

    // 3) RoPE on q,k
    rope_kernel<<<(int)((kM * 32 + 255) / 256), 256, 0, stream>>>(qkv16);

    // 4) window attention (flash, WMMA)
    attn_kernel<<<dim3(kH, kBATCH * 64), 512, 0, stream>>>(qkv16, o16);

    // 5) output projection: [65536,1024] x [1024,1024]^T -> d_out f32
    gemm_wmma<false><<<dim3((int)(kM / 128), kC / 64), 256, 0, stream>>>(
        o16, wproj16, (void*)out, (int)kM, kC, kC);
}
